// KPConvLSTMCell_7842610283201
// MI455X (gfx1250) — matrix operand order repaired and checked by reading, not compiled, not run on hardware
//
#include <hip/hip_runtime.h>
#include <hip/hip_bf16.h>
#include <math.h>

// ---------------------------------------------------------------------------
// KPConv-LSTM cell for MI455X (gfx1250, wave32, WMMA)
// B=4, N=4096, CIN_FEAT=4, FILTERS=64, C_TOT=68, K_POINTS=15, NSAMPLE=16
// ---------------------------------------------------------------------------

#define BATCH     4
#define NPTS      4096
#define CIN       4
#define FILT      64
#define CTOT      68          // FILT + CIN
#define KPOINTS   15
#define NSAMP     16
#define CHANS     67          // 3 + FILT  (row stride of h/c/out)
#define KPAD      1024        // K_POINTS*C_TOT = 1020, padded to 1024
#define MTILE     32          // points per workgroup (2 WMMA M-tiles)
#define AROW      1096        // A-row stride in halves: >= 1088 so the zero
                              // k=15 WMMA row can store unguarded; 16B aligned,
                              // bank stride 548%64=36 -> conflict-free b128

typedef float    v2f  __attribute__((ext_vector_type(2)));
typedef float    v8f  __attribute__((ext_vector_type(8)));
typedef _Float16 v8h  __attribute__((ext_vector_type(8)));
typedef _Float16 v16h __attribute__((ext_vector_type(16)));

__device__ __forceinline__ float hsig(float x) {
    return fminf(fmaxf(0.2f * x + 0.5f, 0.0f), 1.0f);
}

// ---------------------------------------------------------------------------
// Kernel 0: pack W_i/W_f/W_o/W_c (f32, [15][68][64]) into swizzled f16
// B-fragment layout for v_wmma_f32_16x16x32_f16:
//   flat = (((kt*16 + nt)*32 + lane)*16 + j)
//   K = kt*32 + 16*(lane/16) + j          (B: contiguous K per lane group)
//   n = nt*16 + (lane%16),  gate = n/64, f = n%64,  k = K/68, c = K%68
// ---------------------------------------------------------------------------
__global__ __launch_bounds__(256) void packW(const float* __restrict__ Wi,
                                             const float* __restrict__ Wf,
                                             const float* __restrict__ Wo,
                                             const float* __restrict__ Wc,
                                             _Float16* __restrict__ Wsw) {
    int tid = blockIdx.x * 256 + threadIdx.x;       // 0 .. 262143
    int j    = tid & 15;
    int lane = (tid >> 4) & 31;
    int nt   = (tid >> 9) & 15;
    int kt   = tid >> 13;
    int K    = kt * 32 + 16 * (lane >> 4) + j;
    int n    = nt * 16 + (lane & 15);
    int gate = n >> 6;
    int f    = n & 63;
    float v = 0.0f;
    if (K < KPOINTS * CTOT) {
        int k = K / CTOT, c = K % CTOT;
        const float* W = (gate == 0) ? Wi : (gate == 1) ? Wf : (gate == 2) ? Wo : Wc;
        v = W[(k * CTOT + c) * FILT + f];
    }
    Wsw[tid] = (_Float16)v;
}

// ---------------------------------------------------------------------------
// Kernel 1: neighbor search. One thread per query point.
// Sources staged in LDS (stride-4 rows -> one ds_load_b128 broadcast per set).
// Outputs: nn_h, nn_c (argmin), knn idx + squared distances (top-16, ascending).
// ---------------------------------------------------------------------------
#define CHUNK 1024

__global__ __launch_bounds__(256) void knn_kernel(const float* __restrict__ inputs,
                                                  const float* __restrict__ h_tm1,
                                                  const float* __restrict__ c_tm1,
                                                  int*   __restrict__ nn_h,
                                                  int*   __restrict__ nn_c,
                                                  int*   __restrict__ knn,
                                                  float* __restrict__ d2out) {
    __shared__ float Sx[CHUNK][4];
    __shared__ float Sh[CHUNK][4];
    __shared__ float Sc[CHUNK][4];

    int tid = threadIdx.x;
    int b   = blockIdx.x >> 4;                 // 16 blocks per batch
    int n   = ((blockIdx.x & 15) << 8) + tid;  // query index
    long bn = (long)b * NPTS + n;

    float qx = inputs[bn * 7 + 0];
    float qy = inputs[bn * 7 + 1];
    float qz = inputs[bn * 7 + 2];

    float bh_d = 3e38f, bc_d = 3e38f;
    int   bh_i = 0,     bc_i = 0;
    float bd[NSAMP];
    int   bi[NSAMP];
#pragma unroll
    for (int s = 0; s < NSAMP; ++s) { bd[s] = 3e38f; bi[s] = 0; }

    for (int chunk = 0; chunk < NPTS / CHUNK; ++chunk) {
        int jbase = chunk * CHUNK;
        for (int t = tid; t < CHUNK; t += 256) {
            long src = (long)b * NPTS + jbase + t;
            Sx[t][0] = inputs[src * 7 + 0];
            Sx[t][1] = inputs[src * 7 + 1];
            Sx[t][2] = inputs[src * 7 + 2];
            Sh[t][0] = h_tm1[src * CHANS + 0];
            Sh[t][1] = h_tm1[src * CHANS + 1];
            Sh[t][2] = h_tm1[src * CHANS + 2];
            Sc[t][0] = c_tm1[src * CHANS + 0];
            Sc[t][1] = c_tm1[src * CHANS + 1];
            Sc[t][2] = c_tm1[src * CHANS + 2];
        }
        __syncthreads();

        for (int jj = 0; jj < CHUNK; ++jj) {
            int j = jbase + jj;
            float4 px = *(const float4*)&Sx[jj][0];   // one b128 broadcast each
            float4 ph = *(const float4*)&Sh[jj][0];
            float4 pc = *(const float4*)&Sc[jj][0];
            float dx, dy, dz;
            // x <-> x  (top-16)
            dx = qx - px.x; dy = qy - px.y; dz = qz - px.z;
            float d2x = dx * dx + dy * dy + dz * dz;
            if (d2x < bd[NSAMP - 1]) {
                bd[NSAMP - 1] = d2x; bi[NSAMP - 1] = j;
#pragma unroll
                for (int t = NSAMP - 1; t > 0; --t) {
                    if (bd[t] < bd[t - 1]) {
                        float td = bd[t]; bd[t] = bd[t - 1]; bd[t - 1] = td;
                        int   ti = bi[t]; bi[t] = bi[t - 1]; bi[t - 1] = ti;
                    }
                }
            }
            // x <-> h  (argmin, first index wins on tie)
            dx = qx - ph.x; dy = qy - ph.y; dz = qz - ph.z;
            float d2h = dx * dx + dy * dy + dz * dz;
            if (d2h < bh_d) { bh_d = d2h; bh_i = j; }
            // x <-> c
            dx = qx - pc.x; dy = qy - pc.y; dz = qz - pc.z;
            float d2c = dx * dx + dy * dy + dz * dz;
            if (d2c < bc_d) { bc_d = d2c; bc_i = j; }
        }
        __syncthreads();
    }

    nn_h[bn] = bh_i;
    nn_c[bn] = bc_i;
#pragma unroll
    for (int s = 0; s < NSAMP; ++s) {
        knn[bn * NSAMP + s]   = bi[s];
        d2out[bn * NSAMP + s] = bd[s];
    }
}

// ---------------------------------------------------------------------------
// Kernel 2: fused KPConv aggregation + 4-gate GEMM + LSTM epilogue.
// One workgroup (256 threads = 8 waves) per 32-point tile (2 WMMA M-tiles):
// every B fragment load feeds 2 WMMAs and W L2 traffic is halved vs 16-pt.
// ---------------------------------------------------------------------------
__global__ __launch_bounds__(256) void fused_kernel(const float* __restrict__ inputs,
                                                    const float* __restrict__ h_tm1,
                                                    const float* __restrict__ c_tm1,
                                                    const float* __restrict__ kpts,
                                                    const _Float16* __restrict__ Wsw,
                                                    const int*   __restrict__ nn_h,
                                                    const int*   __restrict__ nn_c,
                                                    const int*   __restrict__ knn,
                                                    const float* __restrict__ d2arr,
                                                    float* __restrict__ out) {
    __shared__ _Float16 Abuf[MTILE][AROW];     // A: 32 pts x 1024 K (f16) + pad
    __shared__ union {
        struct {
            _Float16 featS[16][16][80];        // [pt][nbr][chan], zero tail
            float    inflS[16][16][18];        // [pt][kp(16)][nbr], s-stride 18
        } s1;                                  // live: stages 1 / 1.5
        float Gbuf[MTILE][260];                // live: stages 2 / 3
    } ovl;
    __shared__ float kpS[KPOINTS][3];
    __shared__ float cxyz[MTILE][3];

    int tid = threadIdx.x;
    int b   = blockIdx.x / (NPTS / MTILE);
    int m0  = (blockIdx.x % (NPTS / MTILE)) * MTILE;

    int wave = tid >> 5, lane = tid & 31;
    int lm = lane & 15, lg = lane >> 4;

    // ---- stage 0: small staging -------------------------------------------
    if (tid < KPOINTS * 3) kpS[tid / 3][tid % 3] = kpts[tid];
    if (tid < MTILE * 3) {
        cxyz[tid / 3][tid % 3] =
            inputs[((long)b * NPTS + m0 + tid / 3) * 7 + (tid % 3)];
    }
    __syncthreads();

    // ---- stages 1 + 1.5: two 16-point passes reusing featS/inflS ----------
    for (int half = 0; half < 2; ++half) {
        // gather neighbor features + kernel-point influences
        {
            int pl = tid >> 4, s = tid & 15;        // pass-local point, neighbor
            int p  = half * 16 + pl;
            long bn  = (long)b * NPTS + m0 + p;
            int  idx = knn[bn * NSAMP + s];
            float d2v = d2arr[bn * NSAMP + s];
            int  nh  = nn_h[(long)b * NPTS + idx];
            const float* hrow = h_tm1 + ((long)b * NPTS + nh) * CHANS + 3;
#pragma unroll 8
            for (int c = 0; c < FILT; ++c) ovl.s1.featS[pl][s][c] = (_Float16)hrow[c];
            const float* xrow = inputs + ((long)b * NPTS + idx) * 7;
#pragma unroll
            for (int c = 0; c < CIN; ++c) ovl.s1.featS[pl][s][FILT + c] = (_Float16)xrow[3 + c];
#pragma unroll
            for (int c = CTOT; c < 80; ++c) ovl.s1.featS[pl][s][c] = (_Float16)0.0f;

            float rx = xrow[0] - cxyz[p][0];
            float ry = xrow[1] - cxyz[p][1];
            float rz = xrow[2] - cxyz[p][2];
            float mask = (d2v <= 1.0f) ? 1.0f : 0.0f;   // RADIUS^2 = 1
#pragma unroll
            for (int k = 0; k < KPOINTS; ++k) {
                float dx = rx - kpS[k][0], dy = ry - kpS[k][1], dz = rz - kpS[k][2];
                float d = sqrtf(dx * dx + dy * dy + dz * dz + 1e-12f);
                ovl.s1.inflS[pl][k][s] = fmaxf(0.0f, 1.0f - d * (1.0f / 1.2f)) * mask;
            }
            ovl.s1.inflS[pl][15][s] = 0.0f;             // padded row -> zero WMMA row 15
        }
        __syncthreads();

        // weighted[k][c] = sum_s infl[s][k]*feat[s][c] via v_wmma_f32_16x16x4_f32
        for (int pp = 0; pp < 2; ++pp) {
            int pl = wave * 2 + pp;
#pragma unroll
            for (int ct = 0; ct < 5; ++ct) {
                v8f acc = {};
#pragma unroll
                for (int ks = 0; ks < 16; ks += 4) {
                    v2f av = { ovl.s1.inflS[pl][lm][ks + 2 * lg],
                               ovl.s1.inflS[pl][lm][ks + 2 * lg + 1] };
                    v2f bv = { (float)ovl.s1.featS[pl][ks + 2 * lg][ct * 16 + lm],
                               (float)ovl.s1.featS[pl][ks + 2 * lg + 1][ct * 16 + lm] };
                    acc = __builtin_amdgcn_wmma_f32_16x16x4_f32(
                              false, av, false, bv, (short)0, acc, false, false);
                }
                // k=15 row is exactly zero -> store unguarded in k: it lands in
                // cols 1020..1087 (pad), writing the required zeros at 1020..1023.
#pragma unroll
                for (int r = 0; r < 8; ++r) {
                    int kk = r + 8 * lg, cc = ct * 16 + lm;
                    if (cc < CTOT)
                        Abuf[half * 16 + pl][kk * CTOT + cc] = (_Float16)acc[r];
                }
            }
        }
        __syncthreads();   // protects featS/inflS reuse (and Abuf before stage 2)
    }

    // ---- stage 2: gate GEMM (32 x 1024) x (1024 x 256), f16 WMMA ----------
    // wave -> n-tiles {2w, 2w+1}; both 16-pt M-tiles share each B fragment.
    typedef union { v16h v; v8h h[2]; } u16h;
    v8f acc00 = {}, acc01 = {}, acc10 = {}, acc11 = {};
    int nt0 = wave * 2, nt1 = wave * 2 + 1;
#pragma unroll 2
    for (int kt = 0; kt < KPAD / 32; ++kt) {
        int ks = kt * 32;
        u16h a0, a1;
        a0.h[0] = *(const v8h*)&Abuf[lm][ks + 16 * lg];
        a0.h[1] = *(const v8h*)&Abuf[lm][ks + 16 * lg + 8];
        a1.h[0] = *(const v8h*)&Abuf[16 + lm][ks + 16 * lg];
        a1.h[1] = *(const v8h*)&Abuf[16 + lm][ks + 16 * lg + 8];
        const _Float16* bp0 = Wsw + (((kt * 16 + nt0) * 32 + lane) << 4);
        const _Float16* bp1 = Wsw + (((kt * 16 + nt1) * 32 + lane) << 4);
        u16h b0, b1;
        b0.h[0] = *(const v8h*)bp0;  b0.h[1] = *(const v8h*)(bp0 + 8);
        b1.h[0] = *(const v8h*)bp1;  b1.h[1] = *(const v8h*)(bp1 + 8);
        acc00 = __builtin_amdgcn_wmma_f32_16x16x32_f16(
                    false, a0.v, false, b0.v, (short)0, acc00, false, false);
        acc10 = __builtin_amdgcn_wmma_f32_16x16x32_f16(
                    false, a1.v, false, b0.v, (short)0, acc10, false, false);
        acc01 = __builtin_amdgcn_wmma_f32_16x16x32_f16(
                    false, a0.v, false, b1.v, (short)0, acc01, false, false);
        acc11 = __builtin_amdgcn_wmma_f32_16x16x32_f16(
                    false, a1.v, false, b1.v, (short)0, acc11, false, false);
    }
#pragma unroll
    for (int r = 0; r < 8; ++r) {
        ovl.Gbuf[r + 8 * lg][nt0 * 16 + lm]      = acc00[r];
        ovl.Gbuf[r + 8 * lg][nt1 * 16 + lm]      = acc01[r];
        ovl.Gbuf[16 + r + 8 * lg][nt0 * 16 + lm] = acc10[r];
        ovl.Gbuf[16 + r + 8 * lg][nt1 * 16 + lm] = acc11[r];
    }
    __syncthreads();

    // ---- stage 3: LSTM epilogue -------------------------------------------
#pragma unroll
    for (int half = 0; half < 2; ++half) {
        int p = half * 16 + (tid >> 4);
        int u = tid & 15;
        long bn = (long)b * NPTS + m0 + p;
        int nc = nn_c[bn];
        const float* crow = c_tm1 + ((long)b * NPTS + nc) * CHANS + 3;
        float* orow = out + bn * CHANS;
#pragma unroll
        for (int t = 0; t < 4; ++t) {
            int f = t * 16 + u;
            float gi = ovl.Gbuf[p][f];
            float gf = ovl.Gbuf[p][64 + f];
            float go = ovl.Gbuf[p][128 + f];
            float gc = ovl.Gbuf[p][192 + f];
            float iv = hsig(gi), fv = hsig(gf), ov = hsig(go);
            float cn = fv * crow[f] + iv * tanhf(gc);
            orow[3 + f] = ov * tanhf(cn);
        }
        if (u == 0) {
#pragma unroll
            for (int j = 0; j < 3; ++j) {
                float x = cxyz[p][j];
                float s = hsig(x);                 // i = f = o = hsig(x)
                float cn = s * x + s * tanhf(x);   // c_hat_j = x
                orow[j] = s * tanhf(cn);
            }
        }
    }
}

// ---------------------------------------------------------------------------
// Launch
// ---------------------------------------------------------------------------
extern "C" void kernel_launch(void* const* d_in, const int* in_sizes, int n_in,
                              void* d_out, int out_size, void* d_ws, size_t ws_size,
                              hipStream_t stream) {
    const float* inputs = (const float*)d_in[0];
    const float* h_tm1  = (const float*)d_in[1];
    const float* c_tm1  = (const float*)d_in[2];
    const float* Wi     = (const float*)d_in[3];
    const float* Wf     = (const float*)d_in[4];
    const float* Wo     = (const float*)d_in[5];
    const float* Wc     = (const float*)d_in[6];
    const float* kp     = (const float*)d_in[7];
    float* out          = (float*)d_out;

    char* ws = (char*)d_ws;
    const size_t WSW_BYTES = (size_t)32 * 16 * 32 * 16 * sizeof(_Float16); // 512 KB
    _Float16* Wsw  = (_Float16*)ws;
    int*      nn_h = (int*)(ws + WSW_BYTES);
    int*      nn_c = (int*)(ws + WSW_BYTES + (size_t)BATCH * NPTS * 4);
    int*      knn  = (int*)(ws + WSW_BYTES + (size_t)BATCH * NPTS * 8);
    float*    d2   = (float*)(ws + WSW_BYTES + (size_t)BATCH * NPTS * 8
                                  + (size_t)BATCH * NPTS * NSAMP * 4);

    packW<<<(32 * 16 * 32 * 16) / 256, 256, 0, stream>>>(Wi, Wf, Wo, Wc, Wsw);
    knn_kernel<<<BATCH * (NPTS / 256), 256, 0, stream>>>(inputs, h_tm1, c_tm1,
                                                         nn_h, nn_c, knn, d2);
    fused_kernel<<<BATCH * (NPTS / MTILE), 256, 0, stream>>>(inputs, h_tm1, c_tm1,
                                                             kp, Wsw, nn_h, nn_c,
                                                             knn, d2, out);
}
